// AudioCodec_3006477107482
// MI455X (gfx1250) — matrix operand order repaired and checked by reading, compiled, not run on hardware
//
#include <hip/hip_runtime.h>
#include <hip/hip_bf16.h>
#include <math.h>

typedef _Float16 v8h  __attribute__((ext_vector_type(8)));
typedef _Float16 v16h __attribute__((ext_vector_type(16)));
typedef float    v8f  __attribute__((ext_vector_type(8)));

union AFragU { v16h v; v8h h[2]; };

// ---------------------------------------------------------------------------
// Implicit-GEMM conv1d / conv_transpose1d via v_wmma_f32_16x16x32_f16.
//   Apk : packed weights f16, layout [KW][Cout][Cin]
//   X   : activations f16, layout [B=2][Cin][Tin]
//   Y   : f32 output [B=2][Cout][Tout]  (optional bias add)
// Block tile 128(M) x 128(N); 8 wave32 in 4(M) x 2(N), each wave 32x64 via
// 2x4 WMMA 16x16 tiles. Double-buffered LDS; A tile staged with
// GLOBAL_LOAD_ASYNC_TO_LDS_B128 (ASYNCcnt), B tile software-pipelined
// through registers so global latency overlaps the WMMAs.
// transposed=1 implements ConvTranspose1d(k=8,s=2,p=3) (= dilated conv,
// pad 4; kernel flip folded into packing).
// ---------------------------------------------------------------------------
__global__ __launch_bounds__(256)
void conv_wmma(const _Float16* __restrict__ Apk,
               const _Float16* __restrict__ X,
               const float*    __restrict__ bias,
               float*          __restrict__ Y,
               int Cout, int Cin, int KW,
               int Tin, int Tout, int stride, int pad, int transposed)
{
    __shared__ __align__(16) _Float16 As[2][128][40]; // +8 pad: 16B-aligned rows
    __shared__ __align__(16) _Float16 Bs[2][128][40];

    const int tid  = threadIdx.x;
    const int lane = tid & 31;
    const int wave = tid >> 5;
    const int wm   = wave & 3;        // 0..3 -> M offset *32
    const int wn   = wave >> 2;       // 0..1 -> N offset *64
    const int l16  = lane & 15;
    const int hi   = (lane >> 4) & 1;

    const int nBase = blockIdx.x * 128;
    const int oBase = blockIdx.y * 128;
    const size_t MK = (size_t)Cout * Cin;
    const int CC   = Cin >> 5;        // 32-channel chunks
    const int ITER = KW * CC;

    // ---- per-thread loop-invariant B-tile indices (one column per thread)
    const int nloc  = tid & 127;
    const int iBase = tid >> 7;                   // i = iBase + 2*j
    const int col   = nBase + nloc;
    const int bb    = col / Tout;
    const int tt    = col - bb * Tout;

    // ---- per-thread A-tile chunk ids (two 16B chunks of the 128x32 tile)
    const int arow0 = tid >> 2;
    const int aseg0 = (tid & 3) * 8;
    const int arow1 = (tid + 256) >> 2;
    const int aseg1 = ((tid + 256) & 3) * 8;
    const unsigned long long Ab = (unsigned long long)Apk;

    auto stageA_async = [&](int it, int nb) {
        int kk = it / CC;
        int cB = (it - kk * CC) << 5;
        size_t base = ((size_t)kk * MK + (size_t)oBase * Cin + cB) * 2;
        unsigned off0 = (unsigned)(base + ((size_t)arow0 * Cin + aseg0) * 2);
        unsigned off1 = (unsigned)(base + ((size_t)arow1 * Cin + aseg1) * 2);
        unsigned l0 = (unsigned)(unsigned long long)&As[nb][arow0][aseg0];
        unsigned l1 = (unsigned)(unsigned long long)&As[nb][arow1][aseg1];
        asm volatile("global_load_async_to_lds_b128 %0, %1, %2"
                     :: "v"(l0), "v"(off0), "s"(Ab) : "memory");
        asm volatile("global_load_async_to_lds_b128 %0, %1, %2"
                     :: "v"(l1), "v"(off1), "s"(Ab) : "memory");
    };
    auto loadB = [&](int it, _Float16* breg) {
        int kk = it / CC;
        int cB = (it - kk * CC) << 5;
        int tin; bool ok;
        if (transposed) {                          // lhs_dilation=2, pad=4
            int jd = tt + kk - 4;
            ok  = (jd >= 0) && ((jd & 1) == 0);
            tin = jd >> 1;
            ok  = ok && (tin < Tin);
        } else {
            tin = tt * stride - pad + kk;
            ok  = (tin >= 0) && (tin < Tin);
        }
        if (ok) {
            const _Float16* p = X + ((size_t)bb * Cin + (cB + iBase)) * Tin + tin;
            #pragma unroll
            for (int j = 0; j < 16; ++j)
                breg[j] = p[(size_t)(2 * j) * Tin];
        } else {
            #pragma unroll
            for (int j = 0; j < 16; ++j) breg[j] = (_Float16)0.f;
        }
    };
    auto storeB = [&](int nb, const _Float16* breg) {
        #pragma unroll
        for (int j = 0; j < 16; ++j)
            Bs[nb][nloc][iBase + 2 * j] = breg[j];
    };

    v8f acc[2][4] = {};
    _Float16 breg[16];

    // prologue: chunk 0 in flight
    stageA_async(0, 0);
    loadB(0, breg);
    storeB(0, breg);

    for (int it = 0; it < ITER; ++it) {
        int cur = it & 1;
        asm volatile("s_wait_asynccnt 0x0" ::: "memory"); // A(cur) landed in LDS
        __syncthreads();                                  // + B(cur) ds-stores

        if (it + 1 < ITER) {
            stageA_async(it + 1, cur ^ 1);   // async A -> other buffer
            loadB(it + 1, breg);             // B global loads overlap WMMAs
        }

        // ---- fragment loads per CDNA5 16-bit A/B VGPR layouts
        AFragU afr[2], bfr[4];
        #pragma unroll
        for (int mi = 0; mi < 2; ++mi) {
            int row = wm * 32 + mi * 16 + l16;
            int kb  = hi ? 8 : 0;            // lanes>=16: K 8..15 / 24..31
            afr[mi].h[0] = *(const v8h*)&As[cur][row][kb];
            afr[mi].h[1] = *(const v8h*)&As[cur][row][kb + 16];
        }
        #pragma unroll
        for (int ni = 0; ni < 4; ++ni) {
            int coln = wn * 64 + ni * 16 + l16;
            int kb   = hi ? 8 : 0;           // lanes>=16: K 8..15 / 24..31
            bfr[ni].h[0] = *(const v8h*)&Bs[cur][coln][kb];
            bfr[ni].h[1] = *(const v8h*)&Bs[cur][coln][kb + 16];
        }
        #pragma unroll
        for (int mi = 0; mi < 2; ++mi)
            #pragma unroll
            for (int ni = 0; ni < 4; ++ni)
                acc[mi][ni] = __builtin_amdgcn_wmma_f32_16x16x32_f16(
                    false, afr[mi].v, false, bfr[ni].v,
                    (short)0, acc[mi][ni], false, false);

        if (it + 1 < ITER) storeB(cur ^ 1, breg);
    }

    // ---- epilogue: C/D layout (m = r + 8*hi, n = lane&15), coalesced stores
    #pragma unroll
    for (int mi = 0; mi < 2; ++mi) {
        #pragma unroll
        for (int ni = 0; ni < 4; ++ni) {
            int c2 = nBase + wn * 64 + ni * 16 + l16;
            int b  = c2 / Tout;
            int t  = c2 - b * Tout;
            #pragma unroll
            for (int r = 0; r < 8; ++r) {
                int m = oBase + wm * 32 + mi * 16 + r + (hi ? 8 : 0);
                float v = acc[mi][ni][r];
                if (bias) v += bias[m];
                Y[((size_t)b * Cout + m) * Tout + t] = v;
            }
        }
    }
}

// ---------------------------------------------------------------------------
// Weight packing f32 -> f16, dst layout [kk][o][i]; flexible src strides
// (handles normal OIK layout and ConvTranspose IOK with kernel flip).
// ---------------------------------------------------------------------------
__global__ void pack_w(const float* __restrict__ src, _Float16* __restrict__ dst,
                       int M, int Kc, int KW, int so, int si, int sk, int base, int n)
{
    int e = blockIdx.x * 256 + threadIdx.x;
    if (e >= n) return;
    int kk  = e / (M * Kc);
    int rem = e - kk * (M * Kc);
    int o   = rem / Kc;
    int i   = rem - o * Kc;
    long s  = (long)o * so + (long)i * si + (long)kk * sk + base;
    dst[e] = (_Float16)src[s];
}

__global__ void cast_f16k(const float* __restrict__ s, _Float16* __restrict__ d, int n)
{
    int e = blockIdx.x * 256 + threadIdx.x;
    if (e < n) d[e] = (_Float16)s[e];
}

__global__ void zero_f32k(float* p, int n)
{
    int e = blockIdx.x * 256 + threadIdx.x;
    if (e < n) p[e] = 0.f;
}

// ---------------------------------------------------------------------------
// bias + GroupNorm(32) + SiLU (+ optional residual), f32 in -> f16 out.
// One block per (batch, group): channels of a group are contiguous rows.
// ---------------------------------------------------------------------------
__global__ __launch_bounds__(256)
void bias_gn_silu(const float* __restrict__ Yc, const float* __restrict__ bias,
                  const float* __restrict__ gam, const float* __restrict__ bet,
                  const float* __restrict__ res, _Float16* __restrict__ out16,
                  int C, int T, int cpg)
{
    __shared__ float rs[256], rq[256];
    int b  = blockIdx.x >> 5;
    int g  = blockIdx.x & 31;
    int c0 = g * cpg;
    size_t base = ((size_t)b * C + c0) * T;
    int N = cpg * T;
    float s = 0.f, ss = 0.f;
    for (int e = threadIdx.x; e < N; e += 256) {
        float v = Yc[base + e] + bias[c0 + e / T];
        s += v; ss += v * v;
    }
    rs[threadIdx.x] = s; rq[threadIdx.x] = ss;
    __syncthreads();
    for (int st = 128; st > 0; st >>= 1) {
        if (threadIdx.x < st) {
            rs[threadIdx.x] += rs[threadIdx.x + st];
            rq[threadIdx.x] += rq[threadIdx.x + st];
        }
        __syncthreads();
    }
    float mu   = rs[0] / N;
    float var  = rq[0] / N - mu * mu;
    float rsig = rsqrtf(var + 1e-5f);
    for (int e = threadIdx.x; e < N; e += 256) {
        int c   = c0 + e / T;
        float v = Yc[base + e] + bias[c];
        v = (v - mu) * rsig * gam[c] + bet[c];
        float sl = v / (1.f + expf(-v));      // SiLU
        if (res) sl += res[base + e];
        out16[base + e] = (_Float16)sl;
    }
}

// bias add, write f16 (and optionally f32 for the RVQ residual init)
__global__ void bias_add_cast(const float* __restrict__ Y, const float* __restrict__ bias,
                              _Float16* __restrict__ out16, float* __restrict__ out32,
                              int C, int T, int total)
{
    int e = blockIdx.x * 256 + threadIdx.x;
    if (e >= total) return;
    int c = (e / T) % C;
    float v = Y[e] + bias[c];
    if (out32) out32[e] = v;
    out16[e] = (_Float16)v;
}

// tiny boundary convs (2 channels) -- negligible FLOPs, plain VALU
__global__ void enc_in_direct(const float* __restrict__ audio, const float* __restrict__ w,
                              const float* __restrict__ bias, _Float16* __restrict__ out, int T)
{
    int idx = blockIdx.x * 256 + threadIdx.x;
    if (idx >= 2 * 512 * T) return;
    int t = idx % T;
    int o = (idx / T) % 512;
    int b = idx / (T * 512);
    float acc = bias[o];
    #pragma unroll
    for (int i = 0; i < 2; ++i)
        #pragma unroll
        for (int kk = 0; kk < 7; ++kk) {
            int tin = t + kk - 3;
            if (tin >= 0 && tin < T)
                acc += w[(o * 2 + i) * 7 + kk] * audio[(size_t)(b * 2 + i) * T + tin];
        }
    out[idx] = (_Float16)acc;
}

__global__ void dec_out_direct(const _Float16* __restrict__ X, const float* __restrict__ w,
                               const float* __restrict__ bias, float* __restrict__ out, int T)
{
    int idx = blockIdx.x * 256 + threadIdx.x;
    if (idx >= 2 * 2 * T) return;
    int t = idx % T;
    int o = (idx / T) % 2;
    int b = idx / (T * 2);
    float acc = bias[o];
    for (int i = 0; i < 512; ++i) {
        #pragma unroll
        for (int kk = 0; kk < 7; ++kk) {
            int tin = t + kk - 3;
            if (tin >= 0 && tin < T)
                acc += w[(o * 512 + i) * 7 + kk] * (float)X[(size_t)(b * 512 + i) * T + tin];
        }
    }
    out[idx] = tanhf(acc);
}

// ---------------------------------------------------------------------------
// RVQ helpers
// ---------------------------------------------------------------------------
__global__ void cnorm_kernel(const float* __restrict__ cb, float* __restrict__ cn,
                             int rows, int D)
{
    int r = blockIdx.x * 256 + threadIdx.x;
    if (r >= rows) return;
    float s = 0.f;
    const float* p = cb + (size_t)r * D;
    for (int d = 0; d < D; ++d) s += p[d] * p[d];
    cn[r] = s;
}

// argmin over codes == argmax of (r.c - 0.5*||c||^2); S layout [B][2048][T]
__global__ void rvq_argmax(const float* __restrict__ S, const float* __restrict__ cn,
                           int* __restrict__ idxb, float* __restrict__ idxOut,
                           int T, int q)
{
    int col = blockIdx.x * 256 + threadIdx.x;
    if (col >= 2 * T) return;
    int b = col / T;
    int t = col - b * T;
    float best = -3.0e38f;
    int   bi   = 0;
    const float* sp = S + (size_t)b * 2048 * T + t;
    for (int k = 0; k < 2048; ++k) {
        float v = sp[(size_t)k * T] - 0.5f * cn[k];
        if (v > best) { best = v; bi = k; }
    }
    idxb[col] = bi;
    idxOut[(size_t)col * 8 + q] = (float)bi;     // indices output (B,T,Q)
}

// residual update + quantized-sum accumulate + commit/codebook loss reduce
__global__ __launch_bounds__(256)
void rvq_update(float* __restrict__ r32, _Float16* __restrict__ r16,
                float* __restrict__ qsum, const float* __restrict__ cbq,
                const int* __restrict__ idxb, float* __restrict__ lossacc,
                int D, int T)
{
    __shared__ float red[256];
    int e = blockIdx.x * 256 + threadIdx.x;      // total = 2*D*T, exact grid
    int t = e % T;
    int d = (e / T) % D;
    int b = e / (T * D);
    int id = idxb[b * T + t];
    float c  = cbq[(size_t)id * D + d];
    float nr = r32[e] - c;
    r32[e]  = nr;
    r16[e]  = (_Float16)nr;
    qsum[e] += c;
    red[threadIdx.x] = nr * nr;                  // (residual - q)^2
    __syncthreads();
    for (int st = 128; st > 0; st >>= 1) {
        if (threadIdx.x < st) red[threadIdx.x] += red[threadIdx.x + st];
        __syncthreads();
    }
    if (threadIdx.x == 0) atomicAdd(lossacc, red[0]);
}

__global__ void finish_loss(const float* __restrict__ acc, float* __restrict__ out)
{
    // per level loss = 1.25 * mean((r-q)^2); final = mean over 8 levels
    out[0] = acc[0] * (1.25f / (8.0f * 2097152.0f));
}

// ---------------------------------------------------------------------------
extern "C" void kernel_launch(void* const* d_in, const int* in_sizes, int n_in,
                              void* d_out, int out_size, void* d_ws, size_t ws_size,
                              hipStream_t stream)
{
    (void)in_sizes; (void)n_in; (void)out_size; (void)ws_size;
    const float* audio     = (const float*)d_in[0];
    const float* enc_in_w  = (const float*)d_in[1];
    const float* enc_in_b  = (const float*)d_in[2];
    const float *eb_cw[3], *eb_cb[3], *eb_g[3], *eb_bt[3], *eb_rw[3], *eb_rb[3];
    for (int k = 0; k < 3; ++k) {
        eb_cw[k] = (const float*)d_in[3 + 6 * k + 0];
        eb_cb[k] = (const float*)d_in[3 + 6 * k + 1];
        eb_g[k]  = (const float*)d_in[3 + 6 * k + 2];
        eb_bt[k] = (const float*)d_in[3 + 6 * k + 3];
        eb_rw[k] = (const float*)d_in[3 + 6 * k + 4];
        eb_rb[k] = (const float*)d_in[3 + 6 * k + 5];
    }
    const float* enc_out_w = (const float*)d_in[21];
    const float* enc_out_b = (const float*)d_in[22];
    const float* cb        = (const float*)d_in[23];
    const float* dec_in_w  = (const float*)d_in[24];
    const float* dec_in_b  = (const float*)d_in[25];
    const float *db_w[3], *db_b[3], *db_g[3], *db_bt[3];
    for (int k = 0; k < 3; ++k) {
        db_w[k]  = (const float*)d_in[26 + 4 * k + 0];
        db_b[k]  = (const float*)d_in[26 + 4 * k + 1];
        db_g[k]  = (const float*)d_in[26 + 4 * k + 2];
        db_bt[k] = (const float*)d_in[26 + 4 * k + 3];
    }
    const float* dec_out_w = (const float*)d_in[38];
    const float* dec_out_b = (const float*)d_in[39];

    float* out = (float*)d_out;           // [recon 65536][indices 32768][loss 1]
    float* out_idx  = out + 65536;
    float* out_loss = out + 98304;

    // ---- workspace bump allocator
    char* ws = (char*)d_ws;
    size_t off = 0;
    auto alloc = [&](size_t bytes) -> void* {
        void* p = ws + off;
        off += (bytes + 255) & ~(size_t)255;
        return p;
    };
    const size_t BIG = 2ull * 2048 * 4096;            // 16.78M elems
    _Float16* xA16 = (_Float16*)alloc(BIG * 2);
    _Float16* xB16 = (_Float16*)alloc(BIG * 2);
    float* yA      = (float*)alloc(BIG * 4);
    float* yB      = (float*)alloc(BIG * 4);
    float* yR      = (float*)alloc(BIG * 4);
    float* scores  = (float*)alloc(2ull * 2048 * 2048 * 4);
    float* r32     = (float*)alloc(2ull * 512 * 2048 * 4);
    _Float16* r16  = (_Float16*)alloc(2ull * 512 * 2048 * 2);
    float* qsum    = (float*)alloc(2ull * 512 * 2048 * 4);
    int*   idxb    = (int*)alloc(4096 * 4);
    float* cnormv  = (float*)alloc(8 * 2048 * 4);
    float* lossacc = (float*)alloc(256);
    _Float16 *wp_ec[3], *wp_er[3];
    wp_ec[0] = (_Float16*)alloc(7ull * 1024 * 512  * 2);
    wp_er[0] = (_Float16*)alloc(1ull * 1024 * 512  * 2);
    wp_ec[1] = (_Float16*)alloc(7ull * 1024 * 1024 * 2);
    wp_er[1] = (_Float16*)alloc(1ull * 1024 * 1024 * 2);
    wp_ec[2] = (_Float16*)alloc(7ull * 1024 * 1024 * 2);
    wp_er[2] = (_Float16*)alloc(1ull * 1024 * 1024 * 2);
    _Float16* wp_eo = (_Float16*)alloc(3ull * 512  * 1024 * 2);
    _Float16* cb16  = (_Float16*)alloc(8ull * 2048 * 512  * 2);
    _Float16* wp_di = (_Float16*)alloc(3ull * 2048 * 512  * 2);
    _Float16* wp_db[3];
    wp_db[0] = (_Float16*)alloc(8ull * 2048 * 2048 * 2);
    wp_db[1] = (_Float16*)alloc(8ull * 1024 * 2048 * 2);
    wp_db[2] = (_Float16*)alloc(8ull * 512  * 1024 * 2);

    auto nb = [](size_t n) { return dim3((unsigned)((n + 255) / 256)); };
    const dim3 T256(256);

    // ---- pack weights to f16 [kk][o][i]
    auto pack = [&](const float* src, _Float16* dst, int M, int Kc, int KW,
                    int so, int si, int sk, int base) {
        int n = M * Kc * KW;
        hipLaunchKernelGGL(pack_w, nb((size_t)n), T256, 0, stream,
                           src, dst, M, Kc, KW, so, si, sk, base, n);
    };
    const int eC[3] = {512, 1024, 1024};              // block input channels
    for (int k = 0; k < 3; ++k) {
        pack(eb_cw[k], wp_ec[k], 1024, eC[k], 7, eC[k] * 7, 7, 1, 0);
        pack(eb_rw[k], wp_er[k], 1024, eC[k], 1, eC[k],     1, 1, 0);
    }
    pack(enc_out_w, wp_eo, 512,  1024, 3, 1024 * 3, 3, 1, 0);
    pack(dec_in_w,  wp_di, 2048, 512,  3, 512 * 3,  3, 1, 0);
    // ConvTranspose: src (I,O,K); wT[o][i][kk] = w[i][o][7-kk]
    pack(db_w[0], wp_db[0], 2048, 2048, 8, 8, 2048 * 8, -1, 7);
    pack(db_w[1], wp_db[1], 1024, 2048, 8, 8, 1024 * 8, -1, 7);
    pack(db_w[2], wp_db[2], 512,  1024, 8, 8, 512 * 8,  -1, 7);
    hipLaunchKernelGGL(cast_f16k, nb(8ull * 2048 * 512), T256, 0, stream,
                       cb, cb16, 8 * 2048 * 512);

    auto conv = [&](const _Float16* A, const _Float16* X, const float* bias, float* Y,
                    int Cout, int Cin, int KW, int Tin, int Tout,
                    int stride, int pad, int tr) {
        dim3 grid((unsigned)((2 * Tout) / 128), (unsigned)(Cout / 128));
        hipLaunchKernelGGL(conv_wmma, grid, T256, 0, stream,
                           A, X, bias, Y, Cout, Cin, KW, Tin, Tout, stride, pad, tr);
    };
    auto gnorm = [&](const float* Y, const float* bias, const float* g, const float* bt,
                     const float* res, _Float16* o16, int C, int T) {
        hipLaunchKernelGGL(bias_gn_silu, dim3(64), T256, 0, stream,
                           Y, bias, g, bt, res, o16, C, T, C / 32);
    };

    // ================= Encoder =================
    hipLaunchKernelGGL(enc_in_direct, nb(2ull * 512 * 16384), T256, 0, stream,
                       audio, enc_in_w, enc_in_b, xA16, 16384);
    // block 0: 512 -> 1024, T 16384 -> 8192
    conv(wp_er[0], xA16, eb_rb[0], yR, 1024, 512, 1, 16384, 8192, 2, 0, 0);
    conv(wp_ec[0], xA16, nullptr,  yA, 1024, 512, 7, 16384, 8192, 2, 3, 0);
    gnorm(yA, eb_cb[0], eb_g[0], eb_bt[0], yR, xB16, 1024, 8192);
    // block 1: 1024 -> 1024, T 8192 -> 4096
    conv(wp_er[1], xB16, eb_rb[1], yR, 1024, 1024, 1, 8192, 4096, 2, 0, 0);
    conv(wp_ec[1], xB16, nullptr,  yA, 1024, 1024, 7, 8192, 4096, 2, 3, 0);
    gnorm(yA, eb_cb[1], eb_g[1], eb_bt[1], yR, xA16, 1024, 4096);
    // block 2: 1024 -> 1024, T 4096 -> 2048
    conv(wp_er[2], xA16, eb_rb[2], yR, 1024, 1024, 1, 4096, 2048, 2, 0, 0);
    conv(wp_ec[2], xA16, nullptr,  yA, 1024, 1024, 7, 4096, 2048, 2, 3, 0);
    gnorm(yA, eb_cb[2], eb_g[2], eb_bt[2], yR, xB16, 1024, 2048);
    // enc_out: 1024 -> 512, k=3 p=1 -> latent (B,512,2048)
    conv(wp_eo, xB16, nullptr, yA, 512, 1024, 3, 2048, 2048, 1, 1, 0);
    hipLaunchKernelGGL(bias_add_cast, nb(2ull * 512 * 2048), T256, 0, stream,
                       yA, enc_out_b, r16, r32, 512, 2048, 2 * 512 * 2048);

    // ================= RVQ =================
    hipLaunchKernelGGL(zero_f32k, nb(2ull * 512 * 2048), T256, 0, stream,
                       qsum, 2 * 512 * 2048);
    hipLaunchKernelGGL(zero_f32k, dim3(1), T256, 0, stream, lossacc, 1);
    hipLaunchKernelGGL(cnorm_kernel, nb(8ull * 2048), T256, 0, stream,
                       cb, cnormv, 8 * 2048, 512);
    for (int q = 0; q < 8; ++q) {
        // scores[b][k][t] = sum_d cb[q][k][d] * r[b][d][t]  (WMMA GEMM)
        conv(cb16 + (size_t)q * 2048 * 512, r16, nullptr, scores,
             2048, 512, 1, 2048, 2048, 1, 0, 0);
        hipLaunchKernelGGL(rvq_argmax, nb(4096), T256, 0, stream,
                           scores, cnormv + q * 2048, idxb, out_idx, 2048, q);
        hipLaunchKernelGGL(rvq_update, nb(2ull * 512 * 2048), T256, 0, stream,
                           r32, r16, qsum, cb + (size_t)q * 2048 * 512,
                           idxb, lossacc, 512, 2048);
    }
    hipLaunchKernelGGL(finish_loss, dim3(1), dim3(1), 0, stream, lossacc, out_loss);

    // ================= Decoder =================
    hipLaunchKernelGGL(cast_f16k, nb(2ull * 512 * 2048), T256, 0, stream,
                       qsum, xA16, 2 * 512 * 2048);
    conv(wp_di, xA16, nullptr, yA, 2048, 512, 3, 2048, 2048, 1, 1, 0);
    hipLaunchKernelGGL(bias_add_cast, nb(2ull * 2048 * 2048), T256, 0, stream,
                       yA, dec_in_b, xB16, (float*)nullptr, 2048, 2048, 2 * 2048 * 2048);
    // dec block 0: 2048 -> 2048, T 2048 -> 4096
    conv(wp_db[0], xB16, nullptr, yA, 2048, 2048, 8, 2048, 4096, 2, 3, 1);
    gnorm(yA, db_b[0], db_g[0], db_bt[0], nullptr, xA16, 2048, 4096);
    // dec block 1: 2048 -> 1024, T 4096 -> 8192
    conv(wp_db[1], xA16, nullptr, yB, 1024, 2048, 8, 4096, 8192, 2, 3, 1);
    gnorm(yB, db_b[1], db_g[1], db_bt[1], nullptr, xB16, 1024, 8192);
    // dec block 2: 1024 -> 512, T 8192 -> 16384
    conv(wp_db[2], xB16, nullptr, yA, 512, 1024, 8, 8192, 16384, 2, 3, 1);
    gnorm(yA, db_b[2], db_g[2], db_bt[2], nullptr, xA16, 512, 16384);
    // dec_out: 512 -> 2, k=7 p=3, tanh -> recon
    hipLaunchKernelGGL(dec_out_direct, nb(2ull * 2 * 16384), T256, 0, stream,
                       xA16, dec_out_w, dec_out_b, out, 16384);
}
